// NJM_11940009082968
// MI455X (gfx1250) — compile-verified, matrix operand order checked by reading, not run on hardware
//
#include <hip/hip_runtime.h>

// ---------------------------------------------------------------------------
// LayerNorm-LSTM (B=64, T=512, D=H=512, G=2048, L=2) for gfx1250 (MI455X).
// Phase 1 (per layer): xi = LN(x @ W_ih^T + b_ih)  -- bf16 WMMA GEMM + row-LN.
// Phase 2 (per layer): persistent scan, partitioned over BATCH (recurrence is
//   independent per batch row -> NO inter-workgroup sync). 4 WGs x 16 waves,
//   c-state in registers, h in LDS, W_hh streamed from L2 as B-fragments.
// ---------------------------------------------------------------------------

typedef __attribute__((ext_vector_type(16))) __bf16 v16bf;
typedef __attribute__((ext_vector_type(8)))  float  v8f;

#define B_   64
#define T_   512
#define D_   512
#define H_   512
#define G_   2048
#define BT_  (B_ * T_)
#define EPS_ 1e-5f

// ---- float -> bf16 (round to nearest even) --------------------------------
__device__ __forceinline__ unsigned short f2bf(float f) {
  unsigned u = __float_as_uint(f);
  return (unsigned short)((u + 0x7FFFu + ((u >> 16) & 1u)) >> 16);
}
__device__ __forceinline__ unsigned pack2(float a, float b) {
  return (unsigned)f2bf(a) | ((unsigned)f2bf(b) << 16);
}

// ---- bf16 16x32 fragment load (ISA 7.12.2 16-bit A/B layout) --------------
// lane 0-15: row = r0+lane,    K offsets {0..7, 16..23}
// lane16-31: row = r0+lane-16, K offsets {8..15, 24..31}
// -> two contiguous 16-byte loads per lane (compiler merges to *_b128).
__device__ __forceinline__ v16bf frag_ld(const unsigned short* base, int r0,
                                         int k0, int stride, int lane) {
  const unsigned short* p =
      base + (size_t)(r0 + (lane & 15)) * stride + k0 + ((lane >> 4) << 3);
  union { v16bf v; unsigned u[8]; } f;
#pragma unroll
  for (int i = 0; i < 8; ++i) {
    int off = ((i >> 2) << 4) + ((i & 3) << 1);   // elem offs 0,2,4,6,16,18,20,22
    f.u[i] = *(const unsigned*)(p + off);
  }
  return f.v;
}

#define WMMA_BF16(A, Bf, C) \
  __builtin_amdgcn_wmma_f32_16x16x32_bf16(false, (A), false, (Bf), (short)0, (C), false, false)

// ---------------------------------------------------------------------------
// Kernel 1: f32 -> bf16 conversion (vectorized x4)
// ---------------------------------------------------------------------------
__global__ __launch_bounds__(256) void cvt_bf16(const float* __restrict__ s,
                                                unsigned short* __restrict__ d,
                                                int n4) {
  int i = blockIdx.x * 256 + threadIdx.x;
  if (i < n4) {
    float4 v = reinterpret_cast<const float4*>(s)[i];
    uint2 r;
    r.x = pack2(v.x, v.y);
    r.y = pack2(v.z, v.w);
    reinterpret_cast<uint2*>(d)[i] = r;
  }
}

// ---------------------------------------------------------------------------
// Kernel 2: raw = Xb[32768x512] @ Wb^T[2048x512]  (bf16 WMMA, 64x64 tile/WG)
// ---------------------------------------------------------------------------
__global__ __launch_bounds__(256) void gemm_ih(const unsigned short* __restrict__ Xb,
                                               const unsigned short* __restrict__ Wb,
                                               float* __restrict__ raw) {
  __shared__ unsigned short As[64 * 64];   // 8 KB k-panel of X (bf16)
  __shared__ unsigned short Bs[64 * 64];   // 8 KB k-panel of W (bf16)
  const int tid = threadIdx.x;
  const int mb  = blockIdx.x;              // 0..511
  const int nb  = blockIdx.y;              // 0..31
  const int lane = tid & 31, wave = tid >> 5;
  const int mt = wave >> 1;                // 0..3
  const int nt0 = (wave & 1) * 2;          // 0 or 2

  const unsigned short* Ag = Xb + (size_t)mb * 64 * 512;
  const unsigned short* Bg = Wb + (size_t)nb * 64 * 512;

  v8f acc0 = {}, acc1 = {};
#pragma unroll 1
  for (int kp = 0; kp < 512; kp += 64) {
    __syncthreads();
    for (int i = tid; i < 64 * 64 / 8; i += 256) {   // 16-byte copies
      int r = i >> 3, c = (i & 7) * 8;
      *(uint4*)(As + r * 64 + c) = *(const uint4*)(Ag + (size_t)r * 512 + kp + c);
      *(uint4*)(Bs + r * 64 + c) = *(const uint4*)(Bg + (size_t)r * 512 + kp + c);
    }
    __syncthreads();
#pragma unroll
    for (int k = 0; k < 64; k += 32) {
      v16bf a  = frag_ld(As, mt * 16, k, 64, lane);
      v16bf b0 = frag_ld(Bs, nt0 * 16, k, 64, lane);
      v16bf b1 = frag_ld(Bs, (nt0 + 1) * 16, k, 64, lane);
      acc0 = WMMA_BF16(a, b0, acc0);
      acc1 = WMMA_BF16(a, b1, acc1);
    }
  }
  // C/D layout: vgpr r -> M = r + 8*(lane>=16), N = lane&15
  const int n     = lane & 15;
  const int mbase = mb * 64 + mt * 16 + ((lane >> 4) << 3);
  const int ng0   = nb * 64 + nt0 * 16 + n;
#pragma unroll
  for (int r = 0; r < 8; ++r) {
    raw[(size_t)(mbase + r) * G_ + ng0]      = acc0[r];
    raw[(size_t)(mbase + r) * G_ + ng0 + 16] = acc1[r];
  }
}

// ---------------------------------------------------------------------------
// Kernel 3: in-place LayerNorm over G=2048 per row: z = raw+bias; LN(z)*g+b
// ---------------------------------------------------------------------------
__global__ __launch_bounds__(256) void ln2048(float* __restrict__ raw,
                                              const float* __restrict__ bias,
                                              const float* __restrict__ gamma,
                                              const float* __restrict__ beta) {
  __shared__ float rs[256], rq[256];
  __shared__ float sm, srv;
  const int tid = threadIdx.x;
  float* p = raw + (size_t)blockIdx.x * G_;
  float v[8]; float s = 0.f, q = 0.f;
#pragma unroll
  for (int i = 0; i < 8; ++i) {
    int c = tid + i * 256;
    v[i] = p[c] + bias[c];
    s += v[i]; q += v[i] * v[i];
  }
  rs[tid] = s; rq[tid] = q;
  __syncthreads();
  for (int off = 128; off > 0; off >>= 1) {
    if (tid < off) { rs[tid] += rs[tid + off]; rq[tid] += rq[tid + off]; }
    __syncthreads();
  }
  if (tid == 0) {
    float m = rs[0] * (1.f / G_);
    sm = m;
    srv = rsqrtf(fmaxf(rq[0] * (1.f / G_) - m * m, 0.f) + EPS_);
  }
  __syncthreads();
  const float m = sm, rv = srv;
#pragma unroll
  for (int i = 0; i < 8; ++i) {
    int c = tid + i * 256;
    p[c] = (v[i] - m) * rv * gamma[c] + beta[c];
  }
}

// ---------------------------------------------------------------------------
// Kernel 4: persistent LSTM scan. 4 WGs x 512 threads (16 waves).
// WG owns batch rows [16*wg, 16*wg+16). No inter-WG communication.
// Wave w owns N-tiles nt = w + 16*kt (kt=0..7)  => for a hidden column j,
// gates i,f,g,o = acc[kj],acc[2+kj],acc[4+kj],acc[6+kj] in the SAME lane.
// ---------------------------------------------------------------------------
__global__ __launch_bounds__(512) void lstm_scan(
    const float* __restrict__ xi,              // [BT][G] (LN'd input proj)
    const unsigned short* __restrict__ Whh_b,  // [G][H] bf16
    const float* __restrict__ b_hh,            // [G]
    const float* __restrict__ g_hh, const float* __restrict__ be_hh,  // [G]
    const float* __restrict__ g_ho, const float* __restrict__ be_ho,  // [H]
    float* __restrict__ out_seq,               // [B][T][H]
    float* __restrict__ hy, float* __restrict__ cy) {                  // [B][H]
  __shared__ unsigned short hl[16 * 512];      // 16 KB : h state (bf16)
  __shared__ float bhl[G_], ghl[G_], behl[G_]; // 24 KB : gate LN params
  __shared__ float ghol[H_], behol[H_];        //  4 KB : output LN params
  __shared__ float redS[32 * 8], redQ[32 * 8]; //  2 KB : cross-wave partials
  __shared__ float mGs[16], rGs[16], mHs[16], rHs[16];

  const int tid   = threadIdx.x;
  const int wg    = blockIdx.x;           // 0..3
  const int lane  = tid & 31, wave = tid >> 5;   // wave 0..15
  const int n     = lane & 15;
  const int khalf = lane >> 4;
  const int b0    = wg * 16;              // global batch base
  const int bloc0 = khalf * 8;            // local row base for this lane half

  // ---- one-time init -------------------------------------------------------
  for (int i = tid; i < G_; i += 512) { bhl[i] = b_hh[i]; ghl[i] = g_hh[i]; behl[i] = be_hh[i]; }
  for (int i = tid; i < H_; i += 512) { ghol[i] = g_ho[i]; behol[i] = be_ho[i]; }
  for (int i = tid; i < 16 * 512; i += 512) hl[i] = 0;     // h = 0
  float cst[2][8];
#pragma unroll
  for (int kj = 0; kj < 2; ++kj)
#pragma unroll
    for (int r = 0; r < 8; ++r) cst[kj][r] = 0.f;          // c = 0
  __syncthreads();

  for (int t = 0; t < T_; ++t) {
    // prefetch this step's xi tile (global_prefetch_b8), overlaps the GEMM
    for (int p = tid; p < 2048; p += 512) {
      int row = p >> 7, chunk = p & 127;
      __builtin_prefetch(xi + ((size_t)(b0 + row) * T_ + t) * G_ + chunk * 16, 0, 0);
    }

    // ---- GEMM: gates_raw = h[16x512] @ W_hh^T, wave's 8 strided N-tiles ----
    v8f acc[8] = {};
#pragma unroll 4
    for (int k = 0; k < 512; k += 32) {
      v16bf a = frag_ld(hl, 0, k, 512, lane);
#pragma unroll
      for (int kt = 0; kt < 8; ++kt) {
        v16bf b = frag_ld(Whh_b, (wave + 16 * kt) * 16, k, 512, lane);
        acc[kt] = WMMA_BF16(a, b, acc[kt]);
      }
    }

    // ---- + b_hh, per-lane partial sums for LN over G -----------------------
    float s[8] = {}, q[8] = {};
#pragma unroll
    for (int kt = 0; kt < 8; ++kt) {
      float bb = bhl[(wave + 16 * kt) * 16 + n];
#pragma unroll
      for (int r = 0; r < 8; ++r) {
        float z = acc[kt][r] + bb;
        acc[kt][r] = z;
        s[r] += z; q[r] += z * z;
      }
    }
    // butterfly over the 16 lanes of each half (emits ds_permute/swizzle)
#pragma unroll
    for (int r = 0; r < 8; ++r) {
      float sv = s[r], qv = q[r];
#pragma unroll
      for (int msk = 1; msk < 16; msk <<= 1) {
        sv += __shfl_xor(sv, msk, 32);
        qv += __shfl_xor(qv, msk, 32);
      }
      if (n == 0) { redS[(wave * 2 + khalf) * 8 + r] = sv; redQ[(wave * 2 + khalf) * 8 + r] = qv; }
    }
    __syncthreads();
    if (tid < 16) {                        // row stats: 16 wave-contributions
      float S = 0.f, Q = 0.f;
      for (int wv = 0; wv < 16; ++wv) {
        S += redS[(wv * 2 + (tid >> 3)) * 8 + (tid & 7)];
        Q += redQ[(wv * 2 + (tid >> 3)) * 8 + (tid & 7)];
      }
      float m = S * (1.f / G_);
      mGs[tid] = m;
      rGs[tid] = rsqrtf(fmaxf(Q * (1.f / G_) - m * m, 0.f) + EPS_);
    }
    __syncthreads();

    // ---- gates -> cell update (all per-lane, in registers) -----------------
    float mloc[8], rloc[8];
#pragma unroll
    for (int r = 0; r < 8; ++r) { mloc[r] = mGs[bloc0 + r]; rloc[r] = rGs[bloc0 + r]; }
    float sH[8] = {}, qH[8] = {};
#pragma unroll
    for (int kj = 0; kj < 2; ++kj) {
      const int j = (wave + 16 * kj) * 16 + n;          // hidden column
      const float gi0 = ghl[j],        be0 = behl[j];
      const float gi1 = ghl[512 + j],  be1 = behl[512 + j];
      const float gi2 = ghl[1024 + j], be2 = behl[1024 + j];
      const float gi3 = ghl[1536 + j], be3 = behl[1536 + j];
#pragma unroll
      for (int r = 0; r < 8; ++r) {
        const size_t xo = ((size_t)(b0 + bloc0 + r) * T_ + t) * G_;
        float iv = xi[xo + j]          + (acc[kj][r]     - mloc[r]) * rloc[r] * gi0 + be0;
        float fv = xi[xo + 512 + j]    + (acc[2 + kj][r] - mloc[r]) * rloc[r] * gi1 + be1;
        float gv = xi[xo + 1024 + j]   + (acc[4 + kj][r] - mloc[r]) * rloc[r] * gi2 + be2;
        float ov = xi[xo + 1536 + j]   + (acc[6 + kj][r] - mloc[r]) * rloc[r] * gi3 + be3;
        float si = 1.f / (1.f + __expf(-iv));
        float sf = 1.f / (1.f + __expf(-fv));
        float so = 1.f / (1.f + __expf(-ov));
        float cc = sf * cst[kj][r] + si * tanhf(gv);
        cst[kj][r] = cc;
        float tc = tanhf(cc);
        acc[kj][r]     = tc;     // reuse acc registers: tanh(c)
        acc[2 + kj][r] = so;     // reuse acc registers: sigmoid(o)
        sH[r] += tc; qH[r] += tc * tc;
      }
    }
    // ---- LN over H stats ---------------------------------------------------
#pragma unroll
    for (int r = 0; r < 8; ++r) {
      float sv = sH[r], qv = qH[r];
#pragma unroll
      for (int msk = 1; msk < 16; msk <<= 1) {
        sv += __shfl_xor(sv, msk, 32);
        qv += __shfl_xor(qv, msk, 32);
      }
      if (n == 0) { redS[(wave * 2 + khalf) * 8 + r] = sv; redQ[(wave * 2 + khalf) * 8 + r] = qv; }
    }
    __syncthreads();
    if (tid < 16) {
      float S = 0.f, Q = 0.f;
      for (int wv = 0; wv < 16; ++wv) {
        S += redS[(wv * 2 + (tid >> 3)) * 8 + (tid & 7)];
        Q += redQ[(wv * 2 + (tid >> 3)) * 8 + (tid & 7)];
      }
      float m = S * (1.f / H_);
      mHs[tid] = m;
      rHs[tid] = rsqrtf(fmaxf(Q * (1.f / H_) - m * m, 0.f) + EPS_);
    }
    __syncthreads();

    // ---- h = sigmoid(o) * LN(tanh(c)); write LDS h (bf16) + global out -----
    float mh[8], rh[8];
#pragma unroll
    for (int r = 0; r < 8; ++r) { mh[r] = mHs[bloc0 + r]; rh[r] = rHs[bloc0 + r]; }
#pragma unroll
    for (int kj = 0; kj < 2; ++kj) {
      const int j = (wave + 16 * kj) * 16 + n;
      const float gho = ghol[j], beo = behol[j];
#pragma unroll
      for (int r = 0; r < 8; ++r) {
        float tc = acc[kj][r], so = acc[2 + kj][r];
        float h = so * ((tc - mh[r]) * rh[r] * gho + beo);
        hl[(bloc0 + r) * 512 + j] = f2bf(h);
        out_seq[((size_t)(b0 + bloc0 + r) * T_ + t) * H_ + j] = h;
        if (t == T_ - 1) {
          hy[(b0 + bloc0 + r) * 512 + j] = h;
          cy[(b0 + bloc0 + r) * 512 + j] = cst[kj][r];
        }
      }
    }
    __syncthreads();   // h ready for next step's A-fragments
  }
}

// ---------------------------------------------------------------------------
// Host launcher
// ---------------------------------------------------------------------------
extern "C" void kernel_launch(void* const* d_in, const int* in_sizes, int n_in,
                              void* d_out, int out_size, void* d_ws, size_t ws_size,
                              hipStream_t stream) {
  (void)in_sizes; (void)n_in; (void)out_size; (void)ws_size;
  const float* x     = (const float*)d_in[0];
  const float* W_ih  = (const float*)d_in[1];   // [2][2048][512]
  const float* b_ih  = (const float*)d_in[2];   // [2][2048]
  const float* W_hh  = (const float*)d_in[3];   // [2][2048][512]
  const float* b_hh  = (const float*)d_in[4];   // [2][2048]
  const float* g_ih  = (const float*)d_in[5];
  const float* be_ih = (const float*)d_in[6];
  const float* g_hh  = (const float*)d_in[7];
  const float* be_hh = (const float*)d_in[8];
  const float* g_ho  = (const float*)d_in[9];   // [2][512]
  const float* be_ho = (const float*)d_in[10];  // [2][512]

  float* out = (float*)d_out;                    // [64][512][512]
  float* hy  = out + (size_t)B_ * T_ * H_;       // [2][64][512]
  float* cy  = hy + (size_t)2 * B_ * H_;         // [2][64][512]

  char* ws = (char*)d_ws;
  size_t off = 0;
  float* raw = (float*)(ws + off);                     off += (size_t)BT_ * G_ * 4;  // 256 MB (xi in-place)
  unsigned short* xb    = (unsigned short*)(ws + off); off += (size_t)BT_ * D_ * 2;  // 32 MB
  unsigned short* wih_b = (unsigned short*)(ws + off); off += (size_t)G_ * D_ * 2;
  unsigned short* whh_b = (unsigned short*)(ws + off); off += (size_t)G_ * H_ * 2;

  const int nw4 = G_ * D_ / 4;       // weight quarters
  const int nx4 = BT_ * D_ / 4;      // activation quarters
  for (int l = 0; l < 2; ++l) {
    const float* Xl = (l == 0) ? x : out;       // layer 1 reads layer-0 h-seq
    cvt_bf16<<<(nx4 + 255) / 256, 256, 0, stream>>>(Xl, xb, nx4);
    cvt_bf16<<<(nw4 + 255) / 256, 256, 0, stream>>>(W_ih + (size_t)l * G_ * D_, wih_b, nw4);
    cvt_bf16<<<(nw4 + 255) / 256, 256, 0, stream>>>(W_hh + (size_t)l * G_ * H_, whh_b, nw4);
    gemm_ih<<<dim3(BT_ / 64, G_ / 64), 256, 0, stream>>>(xb, wih_b, raw);
    ln2048<<<BT_, 256, 0, stream>>>(raw, b_ih + (size_t)l * G_,
                                    g_ih + (size_t)l * G_, be_ih + (size_t)l * G_);
    lstm_scan<<<4, 512, 0, stream>>>(
        raw, whh_b, b_hh + (size_t)l * G_,
        g_hh + (size_t)l * G_, be_hh + (size_t)l * G_,
        g_ho + (size_t)l * H_, be_ho + (size_t)l * H_,
        out, hy + (size_t)l * B_ * H_, cy + (size_t)l * B_ * H_);
  }
}
